// AvgRnn_17858474017389
// MI455X (gfx1250) — compile-verified, matrix operand order checked by reading, not run on hardware
//
#include <hip/hip_runtime.h>
#include <stdint.h>

// ---------------------------------------------------------------------------
// Shapes (from reference): B=32 S=40 T=60 H=256 F=64 V=50000 OUT=128, D_IN=320
// ---------------------------------------------------------------------------
#define BB   32
#define SS   40
#define TT   60
#define HH   256
#define FF   64
#define DIN  320
#define G4   1024          // 4*H gates per direction
#define NALL 2048          // both directions' gates
#define ROWS 1280          // S*B rows for the G_in GEMM
#define OUTD 128

typedef __attribute__((ext_vector_type(16))) __bf16 v16bf;
typedef __attribute__((ext_vector_type(8)))  float  v8f;

__device__ __forceinline__ unsigned short f2bf(float f) {
  union { float f; uint32_t u; } x; x.f = f;
  uint32_t u = x.u;
  return (unsigned short)((u + 0x7FFFu + ((u >> 16) & 1u)) >> 16);  // RNE
}

// Load one 16x32 bf16 WMMA operand fragment (A layout; B uses the same
// pattern with "row" = output column since weights are stored N-major).
//   lane<16 : K = kbase+{0..7} and kbase+{16..23}
//   lane>=16: K = kbase+{8..15} and kbase+{24..31}
__device__ __forceinline__ v16bf load_frag(const unsigned short* base,
                                           int row_stride, int row,
                                           int kbase, int lane) {
  int khalf = (lane >> 4) * 8;
  const unsigned short* p = base + row * row_stride + kbase + khalf;
  union { uint4 q[2]; v16bf v; } f;
  f.q[0] = *(const uint4*)(p);
  f.q[1] = *(const uint4*)(p + 16);
  return f.v;
}

// ---------------------------------------------------------------------------
// fp32 -> bf16 weight conversion
// ---------------------------------------------------------------------------
__global__ void k_f2bf(const float* __restrict__ src,
                       unsigned short* __restrict__ dst, int n) {
  int i = blockIdx.x * 256 + threadIdx.x;
  if (i < n) dst[i] = f2bf(src[i]);
}

__global__ void k_bias(const float* bihf, const float* bhhf,
                       const float* bihb, const float* bhhb, float* bias) {
  int i = blockIdx.x * 256 + threadIdx.x;
  if (i < G4)       bias[i] = bihf[i] + bhhf[i];
  else if (i < NALL) bias[i] = bihb[i - G4] + bhhb[i - G4];
}

// ---------------------------------------------------------------------------
// Fused embedding gather + token attention + feature sum.
// One 256-thread block per (b,s). Writes bf16 X row at (s*32+b) of size 320.
// ---------------------------------------------------------------------------
__global__ __launch_bounds__(256) void k_sen(
    const int* __restrict__ x, const unsigned char* __restrict__ xmask,
    const float* __restrict__ xfeat, const float* __restrict__ attn_w,
    const float* __restrict__ attn_b, const float* __restrict__ emb,
    unsigned short* __restrict__ Xbf) {
  __shared__ float eL[TT * HH];     // 60 KB: gathered embeddings, reused twice
  __shared__ float score[64];
  __shared__ float alpha[64];
  __shared__ int   vtok[64];
  __shared__ int   idxL[64];
  __shared__ int   validSen;

  int blk = blockIdx.x;             // b*40 + s
  int b = blk / SS, s = blk % SS;
  int tid = threadIdx.x;
  int lane = tid & 31, wave = tid >> 5;
  const int base = (b * SS + s) * TT;

  if (tid < TT) {
    idxL[tid] = x[base + tid];
    vtok[tid] = xmask[base + tid] ? 0 : 1;
  }
  __syncthreads();

  for (int t = 0; t < TT; ++t)
    eL[t * HH + tid] = emb[(size_t)idxL[t] * HH + tid];

  float aw[8];
#pragma unroll
  for (int c = 0; c < 8; ++c) aw[c] = attn_w[lane + 32 * c];
  float ab = attn_b[0];
  __syncthreads();

  // scores: one token per wave iteration (8 waves)
  for (int t = wave; t < TT; t += 8) {
    float p = 0.f;
#pragma unroll
    for (int c = 0; c < 8; ++c) p += eL[t * HH + lane + 32 * c] * aw[c];
#pragma unroll
    for (int off = 16; off > 0; off >>= 1) p += __shfl_xor(p, off, 32);
    if (lane == 0) score[t] = p + ab;
  }
  __syncthreads();

  // masked softmax over T (wave 0)
  if (tid < 32) {
    int v0 = (tid < TT) ? vtok[tid] : 0;
    int v1 = (tid + 32 < TT) ? vtok[tid + 32] : 0;
    float s0 = v0 ? score[tid] : -3.0e38f;
    float s1 = v1 ? score[tid + 32] : -3.0e38f;
    float mx = fmaxf(s0, s1);
    int anyv = v0 | v1;
#pragma unroll
    for (int off = 16; off > 0; off >>= 1) {
      mx = fmaxf(mx, __shfl_xor(mx, off, 32));
      anyv |= __shfl_xor(anyv, off, 32);
    }
    float e0 = v0 ? __expf(s0 - mx) : 0.f;
    float e1 = v1 ? __expf(s1 - mx) : 0.f;
    float sm = e0 + e1;
#pragma unroll
    for (int off = 16; off > 0; off >>= 1) sm += __shfl_xor(sm, off, 32);
    float inv = (sm > 0.f) ? 1.f / sm : 0.f;
    if (tid < TT) alpha[tid] = e0 * inv;
    if (tid + 32 < TT) alpha[tid + 32] = e1 * inv;
    if (tid == 0) validSen = anyv;
  }
  __syncthreads();

  // weighted embedding sum (all 256 threads, one H channel each)
  float acc = 0.f;
  for (int t = 0; t < TT; ++t) acc += alpha[t] * eL[t * HH + tid];
  int vs = validSen;
  unsigned short* row = Xbf + (size_t)(s * BB + b) * DIN;
  row[tid] = f2bf(vs ? acc : 0.f);

  // masked feature sum (threads 0..63)
  if (tid < FF) {
    float f = 0.f;
    const float* fp = xfeat + (size_t)base * FF + tid;
    for (int t = 0; t < TT; ++t)
      if (vtok[t]) f += fp[t * FF];
    row[HH + tid] = f2bf(vs ? f : 0.f);
  }
}

// ---------------------------------------------------------------------------
// G_in = X @ [Wih_f ; Wih_b]^T + (bih + bhh)   -- 1280 x 2048, K = 320
// One 16x16 tile per wave; 4 waves per block; 10240 tiles total.
// ---------------------------------------------------------------------------
__global__ __launch_bounds__(128) void k_gin(
    const unsigned short* __restrict__ Xbf,
    const unsigned short* __restrict__ Wih,
    const float* __restrict__ Bias, float* __restrict__ Gin) {
  int lane = threadIdx.x & 31;
  int wg = blockIdx.x * 4 + (threadIdx.x >> 5);
  int mTile = wg >> 7;          // / 128
  int nTile = wg & 127;
  int col = nTile * 16 + (lane & 15);       // B column == weight row == D col
  int arow = mTile * 16 + (lane & 15);

  float bv = Bias[col];
  v8f acc = { bv, bv, bv, bv, bv, bv, bv, bv };
  for (int k = 0; k < DIN; k += 32) {
    v16bf a = load_frag(Xbf, DIN, arow, k, lane);
    v16bf w = load_frag(Wih, DIN, col, k, lane);
    acc = __builtin_amdgcn_wmma_f32_16x16x32_bf16(false, a, false, w,
                                                  (short)0, acc, false, false);
  }
  int rbase = mTile * 16 + (lane >> 4) * 8;
#pragma unroll
  for (int i = 0; i < 8; ++i)
    Gin[(size_t)(rbase + i) * NALL + col] = acc[i];
}

// ---------------------------------------------------------------------------
// Persistent bidirectional LSTM: 2 blocks (fwd/bwd) x 512 threads (16 waves).
// Wave w owns hidden-column tile w (16 cols) across all 4 gate quadrants and
// both M tiles, so gates combine in-register; c lives in registers, h in LDS.
// ---------------------------------------------------------------------------
__global__ __launch_bounds__(512) void k_lstm(
    const unsigned short* __restrict__ WhhF,
    const unsigned short* __restrict__ WhhB,
    const float* __restrict__ Gin, const int* __restrict__ lens,
    float* __restrict__ Hfin) {
  __shared__ unsigned short h_bf[BB * HH];  // 16 KB: WMMA A feed
  __shared__ float h_f32[BB * HH];          // 32 KB: f32 hidden state
  __shared__ int lensLds[BB];

  int dir = blockIdx.x;
  const unsigned short* Whh = dir ? WhhB : WhhF;
  const float* GinD = Gin + dir * G4;
  int tid = threadIdx.x;
  int lane = tid & 31;
  int wave = tid >> 5;                       // hidden-column tile 0..15
  int ncol = wave * 16 + (lane & 15);        // hidden column for B/D

  for (int i = tid; i < BB * HH; i += 512) { h_bf[i] = 0; h_f32[i] = 0.f; }
  if (tid < BB) lensLds[tid] = lens[tid];
  __syncthreads();

  int lenReg[2][8];
#pragma unroll
  for (int mt = 0; mt < 2; ++mt)
#pragma unroll
    for (int i = 0; i < 8; ++i)
      lenReg[mt][i] = lensLds[mt * 16 + (lane >> 4) * 8 + i];

  v8f cSt[2];
#pragma unroll
  for (int mt = 0; mt < 2; ++mt)
#pragma unroll
    for (int i = 0; i < 8; ++i) cSt[mt][i] = 0.f;

  for (int t = 0; t < SS; ++t) {
    int ss = dir ? (SS - 1 - t) : t;
    v8f hNew[2];
    for (int mt = 0; mt < 2; ++mt) {
      // seed accumulators with precomputed input projection + biases
      v8f g[4];
      int rbase = ss * BB + mt * 16 + (lane >> 4) * 8;
#pragma unroll
      for (int q = 0; q < 4; ++q) {
        int col = q * HH + ncol;
#pragma unroll
        for (int i = 0; i < 8; ++i)
          g[q][i] = GinD[(size_t)(rbase + i) * NALL + col];
      }
      int arow = mt * 16 + (lane & 15);
      for (int k = 0; k < HH; k += 32) {
        v16bf a = load_frag(h_bf, HH, arow, k, lane);
#pragma unroll
        for (int q = 0; q < 4; ++q) {
          v16bf w = load_frag(Whh, HH, q * HH + ncol, k, lane);
          g[q] = __builtin_amdgcn_wmma_f32_16x16x32_bf16(
              false, a, false, w, (short)0, g[q], false, false);
        }
      }
      // gate math fully in registers: same (m,n) slot across the 4 quadrants
#pragma unroll
      for (int i = 0; i < 8; ++i) {
        int bi = mt * 16 + (lane >> 4) * 8 + i;
        bool valid = ss < lenReg[mt][i];
        float gi = 1.f / (1.f + __expf(-g[0][i]));
        float gf = 1.f / (1.f + __expf(-g[1][i]));
        float gc = tanhf(g[2][i]);
        float go = 1.f / (1.f + __expf(-g[3][i]));
        float c2 = gf * cSt[mt][i] + gi * gc;
        float h2 = go * tanhf(c2);
        float hOld = h_f32[bi * HH + ncol];
        cSt[mt][i] = valid ? c2 : cSt[mt][i];
        hNew[mt][i] = valid ? h2 : hOld;
      }
    }
    __syncthreads();   // all waves done reading h for this step
#pragma unroll
    for (int mt = 0; mt < 2; ++mt)
#pragma unroll
      for (int i = 0; i < 8; ++i) {
        int bi = mt * 16 + (lane >> 4) * 8 + i;
        h_f32[bi * HH + ncol] = hNew[mt][i];
        h_bf[bi * HH + ncol] = f2bf(hNew[mt][i]);
      }
    __syncthreads();   // new h visible before next step
  }

  for (int i = tid; i < BB * HH; i += 512)
    Hfin[dir * BB * HH + i] = h_f32[i];
}

// ---------------------------------------------------------------------------
// Head: replicate the reference's stack/reshape interleave, FC, batch-axis
// BN + relu, batch-axis log_softmax. One block, thread j = output column.
// ---------------------------------------------------------------------------
__global__ __launch_bounds__(128) void k_head(
    const float* __restrict__ Hfin, const float* __restrict__ fc_w,
    const float* __restrict__ fc_b, const float* __restrict__ gamma,
    const float* __restrict__ beta, float* __restrict__ out) {
  __shared__ float hid[BB * 512];   // interleaved "hidden" rows
  int tid = threadIdx.x;
  for (int i = tid; i < BB * 512; i += 128) {
    int r = i >> 9, k = i & 511;
    int part = k >> 8, h = k & 255;
    // stack([h_f,h_b],0).reshape(32,512): r<16 -> h_f[2r+part], else h_b[...]
    float v = (r < 16) ? Hfin[(2 * r + part) * HH + h]
                       : Hfin[BB * HH + (2 * (r - 16) + part) * HH + h];
    hid[i] = v;
  }
  __syncthreads();

  int j = tid;
  float logit[32];
#pragma unroll
  for (int r = 0; r < 32; ++r) logit[r] = fc_b[j];
  for (int k = 0; k < 512; ++k) {
    float w = fc_w[j * 512 + k];
#pragma unroll
    for (int r = 0; r < 32; ++r) logit[r] += hid[r * 512 + k] * w;
  }
  float mu = 0.f;
#pragma unroll
  for (int r = 0; r < 32; ++r) mu += logit[r];
  mu *= (1.f / 32.f);
  float var = 0.f;
#pragma unroll
  for (int r = 0; r < 32; ++r) { float d = logit[r] - mu; var += d * d; }
  var *= (1.f / 32.f);
  float inv = rsqrtf(var + 1e-5f);
  float g = gamma[j], be = beta[j];
  float y[32], mx = -3.0e38f;
#pragma unroll
  for (int r = 0; r < 32; ++r) {
    float v = fmaxf(g * (logit[r] - mu) * inv + be, 0.f);
    y[r] = v; mx = fmaxf(mx, v);
  }
  float sm = 0.f;
#pragma unroll
  for (int r = 0; r < 32; ++r) sm += __expf(y[r] - mx);
  float lse = mx + __logf(sm);
#pragma unroll
  for (int r = 0; r < 32; ++r) out[r * OUTD + j] = y[r] - lse;
}

// ---------------------------------------------------------------------------
// Workspace layout (bytes): total ~13.8 MB
// ---------------------------------------------------------------------------
#define OFF_XBF    0u          // 1280*320*2      = 819200
#define OFF_WIH    819200u     // 2048*320*2      = 1310720
#define OFF_WHHF   2129920u    // 1024*256*2      = 524288
#define OFF_WHHB   2654208u    // 1024*256*2      = 524288
#define OFF_BIAS   3178496u    // 2048*4          = 8192
#define OFF_GIN    3186688u    // 1280*2048*4     = 10485760
#define OFF_HFIN   13672448u   // 2*32*256*4      = 65536

extern "C" void kernel_launch(void* const* d_in, const int* in_sizes, int n_in,
                              void* d_out, int out_size, void* d_ws,
                              size_t ws_size, hipStream_t stream) {
  (void)in_sizes; (void)n_in; (void)out_size; (void)ws_size;
  const int*           x      = (const int*)d_in[0];
  const unsigned char* xmask  = (const unsigned char*)d_in[1];
  const float*         xfeat  = (const float*)d_in[2];
  const int*           lens   = (const int*)d_in[3];
  const float*         emb    = (const float*)d_in[6];
  const float*         attn_w = (const float*)d_in[7];
  const float*         attn_b = (const float*)d_in[8];
  const float*         wih_f  = (const float*)d_in[9];
  const float*         whh_f  = (const float*)d_in[10];
  const float*         bih_f  = (const float*)d_in[11];
  const float*         bhh_f  = (const float*)d_in[12];
  const float*         wih_b  = (const float*)d_in[13];
  const float*         whh_b  = (const float*)d_in[14];
  const float*         bih_b  = (const float*)d_in[15];
  const float*         bhh_b  = (const float*)d_in[16];
  const float*         fc_w   = (const float*)d_in[17];
  const float*         fc_b   = (const float*)d_in[18];
  const float*         bn_g   = (const float*)d_in[19];
  const float*         bn_b   = (const float*)d_in[20];

  char* ws = (char*)d_ws;
  unsigned short* Xbf    = (unsigned short*)(ws + OFF_XBF);
  unsigned short* WihAll = (unsigned short*)(ws + OFF_WIH);
  unsigned short* WhhF   = (unsigned short*)(ws + OFF_WHHF);
  unsigned short* WhhB   = (unsigned short*)(ws + OFF_WHHB);
  float*          Bias   = (float*)(ws + OFF_BIAS);
  float*          Gin    = (float*)(ws + OFF_GIN);
  float*          Hfin   = (float*)(ws + OFF_HFIN);

  const int NWIH = G4 * DIN;   // 327680
  const int NWHH = G4 * HH;    // 262144
  k_f2bf<<<(NWIH + 255) / 256, 256, 0, stream>>>(wih_f, WihAll, NWIH);
  k_f2bf<<<(NWIH + 255) / 256, 256, 0, stream>>>(wih_b, WihAll + NWIH, NWIH);
  k_f2bf<<<(NWHH + 255) / 256, 256, 0, stream>>>(whh_f, WhhF, NWHH);
  k_f2bf<<<(NWHH + 255) / 256, 256, 0, stream>>>(whh_b, WhhB, NWHH);
  k_bias<<<NALL / 256, 256, 0, stream>>>(bih_f, bhh_f, bih_b, bhh_b, Bias);

  k_sen<<<BB * SS, 256, 0, stream>>>(x, xmask, xfeat, attn_w, attn_b, emb, Xbf);
  k_gin<<<(ROWS / 16) * (NALL / 16) / 4, 128, 0, stream>>>(Xbf, WihAll, Bias, Gin);
  k_lstm<<<2, 512, 0, stream>>>(WhhF, WhhB, Gin, lens, Hfin);
  k_head<<<1, 128, 0, stream>>>(Hfin, fc_w, fc_b, bn_g, bn_b, (float*)d_out);
}
